// User_83760452206954
// MI455X (gfx1250) — matrix operand order repaired and checked by reading, not compiled, not run on hardware
//
#include <hip/hip_runtime.h>

// ---------------------------------------------------------------------------
// out[2048,128] = mean-pool( A[2048,100000] @ W[128,100000]^T ),  A in {0,1}.
//
// - W split into Whi+Wlo (bf16): two v_wmma_f32_16x16x32_bf16 per 32-K step
//   reproduce f32 precision (A exact in bf16, f32 accumulation).
// - A (819 MB) is the HBM-bound stream (~37 us floor at 23.3 TB/s): staged
//   f32->bf16 into double-buffered LDS once per 16-row M tile; W streams L2.
// - K split 5 ways (100000 = 5*25*800) -> 640 workgroups for HBM saturation.
//   Deterministic: each split writes private partials; fixed-order reduce
//   kernel applies the sum + mean scale.  Row sums are integer-valued floats
//   (A is 0/1) so even LDS float atomics are exact & order-independent.
// ---------------------------------------------------------------------------

typedef __attribute__((ext_vector_type(16))) __bf16 v16bf;
typedef __attribute__((ext_vector_type(8)))  __bf16 v8bf;
typedef __attribute__((ext_vector_type(4)))  __bf16 v4bf;
typedef __attribute__((ext_vector_type(8)))  float  v8f;
typedef __attribute__((ext_vector_type(4)))  float  v4f;

#define BATCH   2048
#define LATENT  128
#define KDIM    100000
#define KSPLIT  5
#define KPS     20000           // K per split = 25 chunks * 800
#define KC      800             // K elements staged per chunk (25 x 32-K WMMA steps)
#define NCHUNK  25              // chunks per K-split
#define LDST    808             // padded LDS row stride (bf16): dword stride 404 ->
                                // bank stride 20 -> 16 rows hit 16 distinct 4-bank groups

// --------------------------- kernel 1: split W -----------------------------
__global__ __launch_bounds__(256)
void split_w_kernel(const v4f* __restrict__ W4,
                    v4bf* __restrict__ hi4,
                    v4bf* __restrict__ lo4,
                    int n4) {
    int i      = blockIdx.x * blockDim.x + threadIdx.x;
    int stride = gridDim.x * blockDim.x;
    for (; i < n4; i += stride) {
        v4f w = W4[i];
        v4bf h, l;
        h[0] = (__bf16)w[0];  l[0] = (__bf16)(w[0] - (float)h[0]);
        h[1] = (__bf16)w[1];  l[1] = (__bf16)(w[1] - (float)h[1]);
        h[2] = (__bf16)w[2];  l[2] = (__bf16)(w[2] - (float)h[2]);
        h[3] = (__bf16)w[3];  l[3] = (__bf16)(w[3] - (float)h[3]);
        hi4[i] = h;
        lo4[i] = l;
    }
}

// --------------------------- kernel 2: partial GEMM ------------------------
// Grid: (128 M-tiles) x (5 K-splits); 256 threads = 8 waves.
// Wave w owns N-tile [16w,16w+16) of a 16x128 partial output tile.
__global__ __launch_bounds__(256)
void gemm_partial_kernel(const float* __restrict__ A,
                         const __bf16* __restrict__ Whi,
                         const __bf16* __restrict__ Wlo,
                         float* __restrict__ partial,   // [KSPLIT][BATCH][LATENT]
                         float* __restrict__ rowsum) {  // [KSPLIT][BATCH]
    __shared__ __align__(16) __bf16 ldsA[2][16 * LDST];
    __shared__ float ldsSum[16];

    const int t     = threadIdx.x;
    const int lane  = t & 31;
    const int wave  = t >> 5;               // 0..7 -> N tile
    const int m0    = blockIdx.x * 16;
    const int n0    = wave * 16;
    const int ksp   = blockIdx.y;           // 0..4
    const int kbase = ksp * KPS;

    if (t < 16) ldsSum[t] = 0.0f;

    // -------- staging: 16 threads per row; 200 float4 per row -> 12..13 each
    const int srow = t >> 4;                // 0..15 : A row within tile
    const int scol = t & 15;                // 0..15 : column-thread within row
    const float* arow = A + (size_t)(m0 + srow) * KDIM + kbase;

    float rsum = 0.0f;

    auto stage = [&](int chunk, int buf) {
        const float* src = arow + chunk * KC;
        __bf16* dst = &ldsA[buf][srow * LDST];
#pragma unroll
        for (int j = 0; j < 13; ++j) {       // float4 index = scol + 16*j (< 200)
            const int f4 = scol + 16 * j;
            if (j < 12 || scol < 8) {
                v4f w = *(const v4f*)(src + f4 * 4);          // global_load_b128
                rsum += (w[0] + w[1]) + (w[2] + w[3]);        // exact: 0/1 values
                v4bf p;
                p[0] = (__bf16)w[0];
                p[1] = (__bf16)w[1];
                p[2] = (__bf16)w[2];
                p[3] = (__bf16)w[3];
                *(v4bf*)(dst + f4 * 4) = p;                   // ds_store_b64
            }
        }
    };

    // -------- WMMA lane mapping (ISA 7.12.2, 16-bit operands, wave32) -------
    const int half = lane >> 4;             // 0 | 1
    const int l16  = lane & 15;
    // B: lane holds column N = n0+l16 (= row of W), 16 contiguous K at +half*16
    const __bf16* bhiRow = Whi + (size_t)(n0 + l16) * KDIM + kbase + half * 16;
    const __bf16* bloRow = Wlo + (size_t)(n0 + l16) * KDIM + kbase + half * 16;

    v8f acc = {};

    stage(0, 0);
    __syncthreads();

    for (int c = 0; c < NCHUNK; ++c) {
        const int buf = c & 1;
        if (c + 1 < NCHUNK) stage(c + 1, buf ^ 1);

        // A: lane holds row M = l16; 16 contiguous K at kk + half*16.
        const __bf16* abase = &ldsA[buf][l16 * LDST + half * 16];
        const __bf16* bh = bhiRow + (size_t)c * KC;
        const __bf16* bl = bloRow + (size_t)c * KC;

#pragma unroll
        for (int kk = 0; kk < KC; kk += 32) {
            v8bf a_lo = *(const v8bf*)(abase + kk);           // 16B LDS load
            v8bf a_hi = *(const v8bf*)(abase + kk + 8);       // 16B LDS load
            v16bf a = __builtin_shufflevector(a_lo, a_hi,
                        0, 1, 2, 3, 4, 5, 6, 7, 8, 9, 10, 11, 12, 13, 14, 15);
            v16bf bvh = *(const v16bf*)(bh + kk);             // 2x global_load_b128
            v16bf bvl = *(const v16bf*)(bl + kk);             // 2x global_load_b128
            acc = __builtin_amdgcn_wmma_f32_16x16x32_bf16(
                      false, a, false, bvh, (short)0, acc, false, false);
            acc = __builtin_amdgcn_wmma_f32_16x16x32_bf16(
                      false, a, false, bvl, (short)0, acc, false, false);
        }
        __syncthreads();
    }

    // -------- per-split row sums (exact integer-valued float adds) ----------
    atomicAdd(&ldsSum[srow], rsum);         // ds_add_f32, order-independent here
    __syncthreads();
    if (t < 16) rowsum[(size_t)ksp * BATCH + m0 + t] = ldsSum[t];

    // D layout: VGPR r holds (M = r + 8*half, N = l16)
    float* pout = partial + ((size_t)ksp * BATCH + m0) * LATENT;
#pragma unroll
    for (int r = 0; r < 8; ++r) {
        const int mloc = r + half * 8;
        pout[(size_t)mloc * LATENT + n0 + l16] = acc[r];
    }
}

// --------------------------- kernel 3: reduce + scale ----------------------
__global__ __launch_bounds__(256)
void reduce_scale_kernel(const float* __restrict__ partial,  // [KSPLIT][BATCH][LATENT]
                         const float* __restrict__ rowsum,   // [KSPLIT][BATCH]
                         float* __restrict__ out) {
    const int idx = blockIdx.x * 256 + threadIdx.x;          // 0 .. BATCH*LATENT-1
    const int b   = idx >> 7;

    float v = 0.0f;
    float s = 0.0f;
#pragma unroll
    for (int k = 0; k < KSPLIT; ++k) {
        v += partial[(size_t)k * BATCH * LATENT + idx];      // fixed order: deterministic
        s += rowsum[(size_t)k * BATCH + b];                  // exact integer adds
    }
    const float sc = (s != 0.0f) ? (1.0f / s) : 1.0f;
    out[idx] = v * sc;
}

// ---------------------------------------------------------------------------
extern "C" void kernel_launch(void* const* d_in, const int* in_sizes, int n_in,
                              void* d_out, int out_size, void* d_ws, size_t ws_size,
                              hipStream_t stream) {
    const float* A = (const float*)d_in[0];   // [2048, 100000] f32 (0/1 values)
    const float* W = (const float*)d_in[1];   // [128, 100000] f32
    float* out     = (float*)d_out;           // [2048, 128] f32

    // Workspace layout (bytes):
    //   [0, 25.6M)          Whi bf16 [128][100000]
    //   [25.6M, 51.2M)      Wlo bf16 [128][100000]
    //   [51.2M, +5.24M)     partial f32 [5][2048][128]
    //   then                rowsum  f32 [5][2048]
    __bf16* whi    = (__bf16*)d_ws;
    __bf16* wlo    = whi + (size_t)LATENT * KDIM;
    float* partial = (float*)(wlo + (size_t)LATENT * KDIM);
    float* rowsum  = partial + (size_t)KSPLIT * BATCH * LATENT;

    const int n4 = (LATENT * KDIM) / 4;       // 3,200,000 float4s
    split_w_kernel<<<(n4 + 255) / 256, 256, 0, stream>>>(
        (const v4f*)W, (v4bf*)whi, (v4bf*)wlo, n4);

    dim3 grid(BATCH / 16, KSPLIT);
    gemm_partial_kernel<<<grid, 256, 0, stream>>>(A, whi, wlo, partial, rowsum);

    reduce_scale_kernel<<<(BATCH * LATENT) / 256, 256, 0, stream>>>(
        partial, rowsum, out);
}
